// SelfAttention_77841987273308
// MI455X (gfx1250) — matrix operand order repaired and checked by reading, not compiled, not run on hardware
//
#include <hip/hip_runtime.h>

// Problem constants (B, M, C, D) from the reference.
#define BB 4
#define MM 4096
#define CC 256
#define DD 64
#define KT 64   // keys processed per attention iteration

typedef __bf16 bf16_t;
typedef __attribute__((ext_vector_type(8)))  __bf16 v8bf;
typedef __attribute__((ext_vector_type(16))) __bf16 v16bf;
typedef __attribute__((ext_vector_type(8)))  float  v8f;

// ---------------------------------------------------------------------------
// WMMA helpers (CDNA5 gfx1250, wave32): D(16x16,f32) = A(16x32,bf16) x B(32x16,bf16) + C
// A-layout per lane: m = lane%16, half = lane/16,
//   elems e=0..15 -> k = k0 + half*16 + e  (contiguous, matches B)
// B-layout per lane: n = lane%16, half = lane/16, elem e -> k = half*16 + e  (contiguous!)
// C/D layout: elem r -> row = r + 8*half, col = lane%16
// ---------------------------------------------------------------------------
static __device__ inline v8f wmma_bf16(v16bf a, v16bf b, v8f c) {
  return __builtin_amdgcn_wmma_f32_16x16x32_bf16(
      /*neg_a=*/false, a, /*neg_b=*/false, b,
      /*c_mod=*/(short)0, c, /*reuse_a=*/false, /*reuse_b=*/false);
}

static __device__ inline v16bf load_a_frag(const bf16_t* row, int k0, int half) {
  v8bf lo = *(const v8bf*)(row + k0 + half * 16);
  v8bf hi = *(const v8bf*)(row + k0 + half * 16 + 8);
  v16bf a;
#pragma unroll
  for (int e = 0; e < 8; ++e) { a[e] = lo[e]; a[e + 8] = hi[e]; }
  return a;
}

static __device__ inline v16bf load_b_frag(const bf16_t* p16) {
  return *(const v16bf*)p16;  // 16 consecutive bf16 (32 B, aligned)
}

static __device__ inline float half_reduce_max(float v) {
#pragma unroll
  for (int m = 1; m < 16; m <<= 1) v = fmaxf(v, __shfl_xor(v, m, 32));
  return v;
}
static __device__ inline float half_reduce_sum(float v) {
#pragma unroll
  for (int m = 1; m < 16; m <<= 1) v += __shfl_xor(v, m, 32);
  return v;
}

// Async global -> LDS copy of one 16-byte chunk per lane (CDNA5 ASYNCcnt path).
static __device__ inline void async_copy_b128(uint32_t lds_addr, uint64_t gaddr) {
  asm volatile("global_load_async_to_lds_b128 %0, %1, off"
               :: "v"(lds_addr), "v"(gaddr) : "memory");
}
static __device__ inline void wait_asynccnt0() {
  asm volatile("s_wait_asynccnt 0" ::: "memory");
}

// ---------------------------------------------------------------------------
// Kernel 0a: fp32 -> bf16 elementwise convert (for p)
// ---------------------------------------------------------------------------
__global__ void cvt_f32_bf16(const float* __restrict__ src, bf16_t* __restrict__ dst, int n) {
  int stride = gridDim.x * blockDim.x;
  for (int i = blockIdx.x * blockDim.x + threadIdx.x; i < n; i += stride)
    dst[i] = (bf16_t)src[i];
}

// ---------------------------------------------------------------------------
// Kernel 0b: weight transpose + convert:  src (K x N, fp32)  ->  dst (N x K, bf16)
// ---------------------------------------------------------------------------
__global__ void transpose_w(const float* __restrict__ src, bf16_t* __restrict__ dst,
                            int K, int N) {
  int idx = blockIdx.x * blockDim.x + threadIdx.x;
  if (idx < K * N) {
    int k = idx / N, n = idx % N;
    dst[n * K + k] = (bf16_t)src[idx];
  }
}

// ---------------------------------------------------------------------------
// Kernel 1: projections. Each wave computes a 16-row tile of:
//   h = p@Wh + bh (row-major bf16), l = p@Wl + bl (row-major bf16),
//   gT[b][c][m] = (p@Wg + bg)[b][m][c]  (transposed bf16).
// ---------------------------------------------------------------------------
__global__ __launch_bounds__(128) void proj_kernel(
    const bf16_t* __restrict__ pb,
    const bf16_t* __restrict__ WhT, const float* __restrict__ bh,
    const bf16_t* __restrict__ WlT, const float* __restrict__ bl,
    const bf16_t* __restrict__ WgT, const float* __restrict__ bg,
    bf16_t* __restrict__ hbuf, bf16_t* __restrict__ lbuf, bf16_t* __restrict__ gT) {
  const int lane = threadIdx.x & 31;
  const int wid  = threadIdx.x >> 5;
  const int tile = blockIdx.x * 4 + wid;           // 16-row tile index, 0..1023
  const int b    = tile / (MM / 16);
  const int row0 = (tile % (MM / 16)) * 16;
  const int half = lane >> 4;
  const int nl   = lane & 15;

  // Cache all 8 A-fragments of the p row block (K = 256 = 8 * 32)
  const bf16_t* prow = pb + (size_t)(b * MM + row0 + nl) * CC;
  v16bf pA[8];
#pragma unroll
  for (int ks = 0; ks < 8; ++ks) pA[ks] = load_a_frag(prow, ks * 32, half);

  const v8f vzero = {0.f, 0.f, 0.f, 0.f, 0.f, 0.f, 0.f, 0.f};

  // ---- h = p @ Wh + bh ---------------------------------------------------
  {
    v8f acc[4] = {vzero, vzero, vzero, vzero};
#pragma unroll
    for (int ks = 0; ks < 8; ++ks)
#pragma unroll
      for (int nt = 0; nt < 4; ++nt) {
        const bf16_t* w = WhT + (size_t)(nt * 16 + nl) * CC + ks * 32 + half * 16;
        acc[nt] = wmma_bf16(pA[ks], load_b_frag(w), acc[nt]);
      }
#pragma unroll
    for (int nt = 0; nt < 4; ++nt) {
      int col = nt * 16 + nl;
      float bias = bh[col];
#pragma unroll
      for (int r = 0; r < 8; ++r) {
        int row = row0 + r + half * 8;
        hbuf[(size_t)(b * MM + row) * DD + col] = (bf16_t)(acc[nt][r] + bias);
      }
    }
  }

  // ---- l = p @ Wl + bl ---------------------------------------------------
  {
    v8f acc[4] = {vzero, vzero, vzero, vzero};
#pragma unroll
    for (int ks = 0; ks < 8; ++ks)
#pragma unroll
      for (int nt = 0; nt < 4; ++nt) {
        const bf16_t* w = WlT + (size_t)(nt * 16 + nl) * CC + ks * 32 + half * 16;
        acc[nt] = wmma_bf16(pA[ks], load_b_frag(w), acc[nt]);
      }
#pragma unroll
    for (int nt = 0; nt < 4; ++nt) {
      int col = nt * 16 + nl;
      float bias = bl[col];
#pragma unroll
      for (int r = 0; r < 8; ++r) {
        int row = row0 + r + half * 8;
        lbuf[(size_t)(b * MM + row) * DD + col] = (bf16_t)(acc[nt][r] + bias);
      }
    }
  }

  // ---- g = p @ Wg + bg, stored transposed gT[b][col][row] ----------------
  for (int cg = 0; cg < 4; ++cg) {  // 4 column groups of 64
    v8f acc[4] = {vzero, vzero, vzero, vzero};
#pragma unroll
    for (int ks = 0; ks < 8; ++ks)
#pragma unroll
      for (int nt = 0; nt < 4; ++nt) {
        const bf16_t* w = WgT + (size_t)(cg * 64 + nt * 16 + nl) * CC + ks * 32 + half * 16;
        acc[nt] = wmma_bf16(pA[ks], load_b_frag(w), acc[nt]);
      }
#pragma unroll
    for (int nt = 0; nt < 4; ++nt) {
      int col = cg * 64 + nt * 16 + nl;
      float bias = bg[col];
#pragma unroll
      for (int r = 0; r < 8; ++r) {
        int row = row0 + r + half * 8;
        gT[(size_t)b * CC * MM + (size_t)col * MM + row] = (bf16_t)(acc[nt][r] + bias);
      }
    }
  }
}

// ---------------------------------------------------------------------------
// Kernel 2: fused flash attention, 64 keys/iteration.
// All 4 waves of a block share the same batch & key loop, so the h tile
// (64x64 bf16) and gT slice (256x64 bf16) are staged ONCE per block into LDS
// with double-buffered async global->LDS copies (ASYNCcnt), then every WMMA
// B-fragment is a contiguous 32B LDS read. Each wave owns one 16-row query
// tile with online softmax; P is transposed (C-layout -> A-layout) via LDS.
// ---------------------------------------------------------------------------
__global__ __launch_bounds__(128) void attn_kernel(
    const bf16_t* __restrict__ hbuf, const bf16_t* __restrict__ lbuf,
    const bf16_t* __restrict__ gT, const float* __restrict__ p,
    float* __restrict__ out) {
  __shared__ bf16_t hs[2][KT * DD];    // 2 x  8 KB : h keys tile (key-major)
  __shared__ bf16_t gs[2][CC * KT];    // 2 x 32 KB : gT slice (col-major rows)
  __shared__ bf16_t ldsP[4][16 * KT];  // 4 x  2 KB : per-wave P tiles

  const int tid  = threadIdx.x;
  const int lane = tid & 31;
  const int wid  = tid >> 5;
  const int qt   = blockIdx.x * 4 + wid;          // query tile, 0..1023
  const int b    = qt / (MM / 16);
  const int qrow0 = (qt % (MM / 16)) * 16;
  const int half = lane >> 4;
  const int nl   = lane & 15;

  const bf16_t* hbase = hbuf + (size_t)b * MM * DD;
  const bf16_t* gbase = gT + (size_t)b * CC * MM;

  // Issue the async copies for one 64-key slice into buffer `buf`.
  auto issue_copy = [&](int buf, int kb) {
    // gT slice: 256 cols x 64 keys = 32 KB = 2048 x 16B chunks, 16 per thread
#pragma unroll
    for (int i = 0; i < 16; ++i) {
      int u = i * 128 + tid;
      int col = u >> 3, ch = u & 7;                // 8 chunks per 128B col-row
      uint64_t ga = (uint64_t)(uintptr_t)(gbase + (size_t)col * MM + kb + ch * 8);
      uint32_t la = (uint32_t)(uintptr_t)&gs[buf][col * KT + ch * 8];
      async_copy_b128(la, ga);
    }
    // h tile: 64 keys x 64 d = 8 KB = 512 x 16B chunks, 4 per thread
#pragma unroll
    for (int i = 0; i < 4; ++i) {
      int u = i * 128 + tid;
      int row = u >> 3, ch = u & 7;
      uint64_t ga = (uint64_t)(uintptr_t)(hbase + (size_t)(kb + row) * DD + ch * 8);
      uint32_t la = (uint32_t)(uintptr_t)&hs[buf][row * DD + ch * 8];
      async_copy_b128(la, ga);
    }
  };

  // Resident A-fragments of l for this query tile (K = D = 64 -> 2 frags)
  const bf16_t* lrow = lbuf + (size_t)(b * MM + qrow0 + nl) * DD;
  const v16bf lA0 = load_a_frag(lrow, 0, half);
  const v16bf lA1 = load_a_frag(lrow, 32, half);

  const v8f vzero = {0.f, 0.f, 0.f, 0.f, 0.f, 0.f, 0.f, 0.f};
  v8f acc[16];
#pragma unroll
  for (int t = 0; t < 16; ++t) acc[t] = vzero;
  float rmax[8], rsum[8];
#pragma unroll
  for (int r = 0; r < 8; ++r) { rmax[r] = -3.0e38f; rsum[r] = 0.f; }

  bf16_t* myP = ldsP[wid];

  issue_copy(0, 0);  // prologue: fill buffer 0

  for (int kb = 0; kb < MM; kb += KT) {
    const int buf = (kb / KT) & 1;
    wait_asynccnt0();     // this wave's async copies have landed in LDS
    __syncthreads();      // every wave's copies have landed
    if (kb + KT < MM) issue_copy(buf ^ 1, kb + KT);  // overlap next slice

    // ---- scores S(16x64) = l(16x64) @ h^T (from LDS) -------------------
    v8f s[4];
#pragma unroll
    for (int sub = 0; sub < 4; ++sub) {
      const bf16_t* hrow = &hs[buf][(sub * 16 + nl) * DD];
      v8f sv = vzero;
      sv = wmma_bf16(lA0, load_b_frag(hrow + half * 16), sv);
      sv = wmma_bf16(lA1, load_b_frag(hrow + 32 + half * 16), sv);
      s[sub] = sv;
    }

    // ---- online softmax (per output row r + 8*half) --------------------
#pragma unroll
    for (int r = 0; r < 8; ++r) {
      float cur = fmaxf(fmaxf(s[0][r], s[1][r]), fmaxf(s[2][r], s[3][r]));
      cur = half_reduce_max(cur);
      float mnew = fmaxf(rmax[r], cur);
      float scale = __expf(rmax[r] - mnew);
      float p0 = __expf(s[0][r] - mnew);
      float p1 = __expf(s[1][r] - mnew);
      float p2 = __expf(s[2][r] - mnew);
      float p3 = __expf(s[3][r] - mnew);
      float ps = half_reduce_sum((p0 + p1) + (p2 + p3));
      rsum[r] = rsum[r] * scale + ps;
      rmax[r] = mnew;
#pragma unroll
      for (int t = 0; t < 16; ++t) acc[t][r] *= scale;
      int row = r + half * 8;
      myP[row * KT + nl]      = (bf16_t)p0;
      myP[row * KT + 16 + nl] = (bf16_t)p1;
      myP[row * KT + 32 + nl] = (bf16_t)p2;
      myP[row * KT + 48 + nl] = (bf16_t)p3;
    }
    __syncthreads();  // order P stores vs cross-lane P loads (uniform barrier)

    // ---- out(16x256) += P(16x64) @ g(64x256) (both from LDS) -----------
    v16bf pA0 = load_a_frag(myP + nl * KT, 0, half);
    v16bf pA1 = load_a_frag(myP + nl * KT, 32, half);
#pragma unroll
    for (int t = 0; t < 16; ++t) {
      const bf16_t* gcol = &gs[buf][(t * 16 + nl) * KT];
      acc[t] = wmma_bf16(pA0, load_b_frag(gcol + half * 16), acc[t]);
      acc[t] = wmma_bf16(pA1, load_b_frag(gcol + 32 + half * 16), acc[t]);
    }
  }

  // ---- epilogue: out = p + attn@g (normalize by row sums) ---------------
#pragma unroll
  for (int t = 0; t < 16; ++t) {
    int col = t * 16 + nl;
#pragma unroll
    for (int r = 0; r < 8; ++r) {
      int row = qrow0 + r + half * 8;
      size_t idx = (size_t)(b * MM + row) * CC + col;
      out[idx] = p[idx] + acc[t][r] / rsum[r];
    }
  }
}

// ---------------------------------------------------------------------------
// Host launcher
// ---------------------------------------------------------------------------
extern "C" void kernel_launch(void* const* d_in, const int* in_sizes, int n_in,
                              void* d_out, int out_size, void* d_ws, size_t ws_size,
                              hipStream_t stream) {
  const float* p  = (const float*)d_in[0];
  const float* Wh = (const float*)d_in[1];
  const float* bh = (const float*)d_in[2];
  const float* Wl = (const float*)d_in[3];
  const float* bl = (const float*)d_in[4];
  const float* Wg = (const float*)d_in[5];
  const float* bg = (const float*)d_in[6];
  float* out = (float*)d_out;

  // Workspace carve-out (bf16 intermediates), 256-byte aligned slices.
  char* ws = (char*)d_ws;
  size_t off = 0;
  auto carve = [&](size_t bytes) {
    char* ptr = ws + off;
    off = (off + bytes + 255) & ~(size_t)255;
    return ptr;
  };
  bf16_t* pb   = (bf16_t*)carve((size_t)BB * MM * CC * 2);  // p in bf16
  bf16_t* hbuf = (bf16_t*)carve((size_t)BB * MM * DD * 2);  // h row-major
  bf16_t* lbuf = (bf16_t*)carve((size_t)BB * MM * DD * 2);  // l row-major
  bf16_t* gTb  = (bf16_t*)carve((size_t)BB * CC * MM * 2);  // g transposed
  bf16_t* WhT  = (bf16_t*)carve((size_t)DD * CC * 2);
  bf16_t* WlT  = (bf16_t*)carve((size_t)DD * CC * 2);
  bf16_t* WgT  = (bf16_t*)carve((size_t)CC * CC * 2);

  // Stage 0: precision conversion + weight transposes.
  {
    int n = BB * MM * CC;
    cvt_f32_bf16<<<2048, 256, 0, stream>>>(p, pb, n);
    transpose_w<<<(CC * DD + 255) / 256, 256, 0, stream>>>(Wh, WhT, CC, DD);
    transpose_w<<<(CC * DD + 255) / 256, 256, 0, stream>>>(Wl, WlT, CC, DD);
    transpose_w<<<(CC * CC + 255) / 256, 256, 0, stream>>>(Wg, WgT, CC, CC);
  }

  // Stage 1: h/l/g projections (WMMA bf16).
  proj_kernel<<<(BB * MM / 16) / 4, 128, 0, stream>>>(pb, WhT, bh, WlT, bl, WgT, bg,
                                                      hbuf, lbuf, gTb);

  // Stage 2: fused flash attention + residual (WMMA bf16, async LDS staging).
  attn_kernel<<<(BB * MM / 16) / 4, 128, 0, stream>>>(hbuf, lbuf, gTb, p, out);
}
